// SimpleLSTM_48069273977364
// MI455X (gfx1250) — matrix operand order repaired and checked by reading, not compiled, not run on hardware
//
#include <hip/hip_runtime.h>
#include <math.h>

typedef __bf16 bf16;
typedef __attribute__((ext_vector_type(16))) __bf16 v16bf;
typedef __attribute__((ext_vector_type(8)))  __bf16 v8bf;
typedef __attribute__((ext_vector_type(8)))  float  v8f;

#define NB    64      // batch
#define SEQ   512     // timesteps
#define HID   1024
#define EMB   1024
#define KTOT  2048    // HID + EMB
#define GATES 4096    // 4*HID

// ---------------------------------------------------------------------------
// Kernel 1: weights fp32 -> bf16 (layout unchanged [4096][2048];
// cols [0,1024) multiply H_t, cols [1024,2048) multiply x_t).
// ---------------------------------------------------------------------------
__global__ void wconvert_kernel(const float* __restrict__ w,
                                bf16* __restrict__ out, int n) {
    int i = blockIdx.x * blockDim.x + threadIdx.x;
    int stride = gridDim.x * blockDim.x;
    for (; i < n; i += stride) out[i] = (bf16)w[i];
}

// ---------------------------------------------------------------------------
// Kernel 2: embedding gather + fp32 -> bf16, Xe layout [t][n][k].
// ---------------------------------------------------------------------------
__global__ void embed_kernel(const int* __restrict__ X,
                             const float* __restrict__ E,
                             bf16* __restrict__ Xe) {
    int b = blockIdx.x;          // b = t*64 + n
    int t = b >> 6;
    int n = b & 63;
    int row = X[n * SEQ + t];    // X is (N, L) row-major
    const float* src = E + (size_t)row * EMB;
    bf16* dst = Xe + (size_t)b * EMB;
    for (int k = threadIdx.x; k < EMB; k += blockDim.x)
        dst[k] = (bf16)src[k];
}

// ---------------------------------------------------------------------------
// Kernel 3: zero-init recurrent state.
// ---------------------------------------------------------------------------
__global__ void init_state_kernel(float* __restrict__ C,
                                  bf16* __restrict__ h0,
                                  bf16* __restrict__ h1) {
    int i = blockIdx.x * blockDim.x + threadIdx.x;
    if (i < NB * HID) {
        C[i] = 0.0f;
        h0[i] = (bf16)0.0f;
        h1[i] = (bf16)0.0f;
    }
}

// ---------------------------------------------------------------------------
// Kernel 4: input projection (no recurrent dependency, fully parallel):
//   Gx[t][n][j] = bias[j] + sum_k Xe[t][n][k] * W[j][1024+k]
// Output stored tile-swizzled to match WMMA C/D lane layout:
//   float Gx[ ((t*4 + mtile)*256 + jtile) * 256 + lane*8 + r ]
// so both the store here and the accumulator-init load in the step kernel
// are single contiguous 32-byte v8f accesses per lane.
// Grid: (bn=64, t=512), 256 threads = 8 waves, each wave 2 tiles, K=1024.
// ---------------------------------------------------------------------------
__launch_bounds__(256, 2)
__global__ void inproj_kernel(const bf16* __restrict__ Xe,   // [512*64][1024]
                              const bf16* __restrict__ W,    // [4096][2048]
                              const float* __restrict__ bias,
                              float* __restrict__ Gx) {
    __shared__ __align__(16) bf16 sA[64][128];   // 16 KB

    const int tid   = threadIdx.x;
    const int wave  = tid >> 5;
    const int lane  = tid & 31;
    const int bn    = blockIdx.x;        // 64 j-columns per block
    const int t     = blockIdx.y;

    const int mtile = wave & 3;          // 16 batch rows
    const int jp    = (wave >> 2) * 2;
    const int jt0   = bn * 4 + jp;
    const int jt1   = jt0 + 1;

    const int mrow  = lane & 15;
    const int khalf = lane >> 4;
    const int j0    = jt0 * 16 + mrow;
    const int j1    = jt1 * 16 + mrow;

    const float b0s = bias[j0];
    const float b1s = bias[j1];
    v8f acc0 = {b0s, b0s, b0s, b0s, b0s, b0s, b0s, b0s};
    v8f acc1 = {b1s, b1s, b1s, b1s, b1s, b1s, b1s, b1s};

    for (int kb = 0; kb < HID; kb += 128) {
        const bf16* Abase = Xe + ((size_t)t * 64) * EMB + kb;
#pragma unroll
        for (int it = 0; it < 4; ++it) {
            int c   = tid + it * 256;        // 1024 chunks of 8 bf16
            int r   = c >> 4;
            int col = (c & 15) * 8;
            *(uint4*)&sA[r][col] = *(const uint4*)(Abase + (size_t)r * EMB + col);
        }
        __syncthreads();
#pragma unroll
        for (int kk = 0; kk < 128; kk += 32) {
            const bf16* ap = &sA[mtile * 16 + mrow][kk + khalf * 16];
            v8bf alo = *(const v8bf*)ap;
            v8bf ahi = *(const v8bf*)(ap + 8);
            v16bf a = __builtin_shufflevector(alo, ahi,
                        0,1,2,3,4,5,6,7,8,9,10,11,12,13,14,15);
            const size_t kcol = (size_t)(HID + kb + kk + khalf * 16);
            v16bf b0 = *(const v16bf*)(W + (size_t)j0 * KTOT + kcol);
            v16bf b1 = *(const v16bf*)(W + (size_t)j1 * KTOT + kcol);
            acc0 = __builtin_amdgcn_wmma_f32_16x16x32_bf16(
                       false, a, false, b0, (short)0, acc0, false, false);
            acc1 = __builtin_amdgcn_wmma_f32_16x16x32_bf16(
                       false, a, false, b1, (short)0, acc1, false, false);
        }
        __syncthreads();
    }

    const size_t tb = ((size_t)t * 4 + mtile) * 256;
    *(v8f*)(Gx + (tb + jt0) * 256 + lane * 8) = acc0;
    *(v8f*)(Gx + (tb + jt1) * 256 + lane * 8) = acc1;
}

// ---------------------------------------------------------------------------
// Kernel 5: one LSTM timestep (critical path).
// Grid: 128 WGs of 128 threads (4 waves): blockIdx -> 16 hidden cols x 32
// batch rows; wave w -> mtile lm = w&1, gates g0 = (w>>1)*2 (2 tiles/wave).
// USE_GX=true : K=1024 (H only), accumulators initialized from swizzled Gx
//               (bias already folded in).
// USE_GX=false: fused K=2048 over concat(H, x_t), bias added at the end.
// ---------------------------------------------------------------------------
template <bool USE_GX>
__launch_bounds__(128)
__global__ void lstm_step_kernel(const bf16* __restrict__ Hin,   // [64][1024]
                                 const bf16* __restrict__ Xt,    // [64][1024]
                                 const bf16* __restrict__ W,     // [4096][2048]
                                 const float* __restrict__ bias, // [4096]
                                 const float* __restrict__ Gxt,  // swizzled, at t
                                 float* __restrict__ C,          // [64][1024]
                                 bf16* __restrict__ Hout_bf,     // [64][1024]
                                 float* __restrict__ Hout_f) {   // [64][1024]
    __shared__ __align__(16) bf16 sA[32][128];   // 8 KB A staging
    __shared__ float sG[4][32][16];              // 8 KB gate staging

    const int tid   = threadIdx.x;
    const int wave  = tid >> 5;          // 0..3
    const int lane  = tid & 31;
    const int h0    = (blockIdx.x >> 1) * 16;    // hidden column group
    const int r0    = (blockIdx.x & 1) * 32;     // batch row group

    const int lm    = wave & 1;          // local 16-row tile
    const int g0    = (wave >> 1) * 2;   // gates {0,1} or {2,3}
    const int mrow  = lane & 15;
    const int khalf = lane >> 4;

    v8f acc0 = {0.f, 0.f, 0.f, 0.f, 0.f, 0.f, 0.f, 0.f};
    v8f acc1 = acc0;
    if (USE_GX) {
        const size_t mg  = (size_t)((r0 >> 4) + lm);      // global row tile 0..3
        const size_t jt0 = (size_t)(g0)       * 64 + (h0 >> 4);
        const size_t jt1 = (size_t)(g0 + 1)   * 64 + (h0 >> 4);
        acc0 = *(const v8f*)(Gxt + (mg * 256 + jt0) * 256 + lane * 8);
        acc1 = *(const v8f*)(Gxt + (mg * 256 + jt1) * 256 + lane * 8);
    }

    const int KLOOP = USE_GX ? HID : KTOT;
    for (int kb = 0; kb < KLOOP; kb += 128) {
        const bf16* Abase = (kb < HID)
            ? (Hin + (size_t)r0 * HID + kb)
            : (Xt  + (size_t)r0 * HID + (kb - HID));
#pragma unroll
        for (int it = 0; it < 4; ++it) {
            int c   = tid + it * 128;        // 512 chunks of 8 bf16
            int r   = c >> 4;                // 0..31
            int col = (c & 15) * 8;
            *(uint4*)&sA[r][col] = *(const uint4*)(Abase + (size_t)r * HID + col);
        }
        __syncthreads();
#pragma unroll
        for (int kk = 0; kk < 128; kk += 32) {
            const bf16* ap = &sA[lm * 16 + mrow][kk + khalf * 16];
            v8bf alo = *(const v8bf*)ap;
            v8bf ahi = *(const v8bf*)(ap + 8);
            v16bf a = __builtin_shufflevector(alo, ahi,
                        0,1,2,3,4,5,6,7,8,9,10,11,12,13,14,15);
            const size_t kcol = (size_t)(kb + kk + khalf * 16);
            const bf16* bp0 = W + (size_t)((g0    ) * HID + h0 + mrow) * KTOT + kcol;
            const bf16* bp1 = W + (size_t)((g0 + 1) * HID + h0 + mrow) * KTOT + kcol;
            v16bf b0 = *(const v16bf*)bp0;
            v16bf b1 = *(const v16bf*)bp1;
            acc0 = __builtin_amdgcn_wmma_f32_16x16x32_bf16(
                       false, a, false, b0, (short)0, acc0, false, false);
            acc1 = __builtin_amdgcn_wmma_f32_16x16x32_bf16(
                       false, a, false, b1, (short)0, acc1, false, false);
        }
        __syncthreads();
    }

    // Spill D tiles to LDS: lane holds rows (khalf*8 + r), col mrow.
#pragma unroll
    for (int r = 0; r < 8; ++r) {
        int m = lm * 16 + khalf * 8 + r;
        sG[g0    ][m][mrow] = acc0[r];
        sG[g0 + 1][m][mrow] = acc1[r];
    }
    __syncthreads();

    // Elementwise LSTM cell update: 32x16 elems, 4 per thread.
    {
        int hl = tid & 15;
        int n0 = (tid >> 4) * 4;
        float bi = 0.f, bf_ = 0.f, bo = 0.f, bg = 0.f;
        if (!USE_GX) {
            bi  = bias[0 * HID + h0 + hl];
            bf_ = bias[1 * HID + h0 + hl];
            bo  = bias[2 * HID + h0 + hl];
            bg  = bias[3 * HID + h0 + hl];
        }
#pragma unroll
        for (int q = 0; q < 4; ++q) {
            int nl = n0 + q;             // local row
            int n  = r0 + nl;            // global batch row
            float gi = sG[0][nl][hl] + bi;   // input gate
            float gf = sG[1][nl][hl] + bf_;  // forget gate
            float go = sG[2][nl][hl] + bo;   // output gate
            float gg = sG[3][nl][hl] + bg;   // cell candidate
            float i_ = 1.0f / (1.0f + __expf(-gi));
            float f_ = 1.0f / (1.0f + __expf(-gf));
            float o_ = 1.0f / (1.0f + __expf(-go));
            float g_ = tanhf(gg);
            size_t idx = (size_t)n * HID + h0 + hl;
            float c_ = f_ * C[idx] + i_ * g_;
            C[idx] = c_;
            float h_ = o_ * tanhf(c_);
            Hout_bf[idx] = (bf16)h_;
            Hout_f[idx]  = h_;
        }
    }
}

// ---------------------------------------------------------------------------
// Host launcher. ws layout (bytes):
//   Wcat bf16 [4096][2048]            16 MB
//   Xe   bf16 [512][64][1024]         64 MB
//   C    fp32 [64][1024]             256 KB
//   Hbf0/Hbf1 bf16 [64][1024]        128 KB each
//   Gx   fp32 swizzled [512*4*256*256]  512 MB   (only if ws_size allows)
// ---------------------------------------------------------------------------
extern "C" void kernel_launch(void* const* d_in, const int* in_sizes, int n_in,
                              void* d_out, int out_size, void* d_ws, size_t ws_size,
                              hipStream_t stream) {
    const int*   X  = (const int*)d_in[0];
    const float* E  = (const float*)d_in[1];
    const float* Ww = (const float*)d_in[2];
    const float* Wb = (const float*)d_in[3];
    float* out = (float*)d_out;

    char* ws = (char*)d_ws;
    size_t off = 0;
    bf16* Wcat = (bf16*)(ws + off); off += (size_t)GATES * KTOT * sizeof(bf16);
    bf16* Xe   = (bf16*)(ws + off); off += (size_t)SEQ * NB * EMB * sizeof(bf16);
    float* C   = (float*)(ws + off); off += (size_t)NB * HID * sizeof(float);
    bf16* Hb0  = (bf16*)(ws + off); off += (size_t)NB * HID * sizeof(bf16);
    bf16* Hb1  = (bf16*)(ws + off); off += (size_t)NB * HID * sizeof(bf16);
    float* Gx  = (float*)(ws + off);
    const size_t need_gx = off + (size_t)SEQ * NB * GATES * sizeof(float);
    const bool use_gx = (ws_size >= need_gx);
    (void)in_sizes; (void)n_in; (void)out_size;

    wconvert_kernel<<<4096, 256, 0, stream>>>(Ww, Wcat, GATES * KTOT);
    embed_kernel<<<SEQ * NB, 256, 0, stream>>>(X, E, Xe);
    init_state_kernel<<<(NB * HID + 255) / 256, 256, 0, stream>>>(C, Hb0, Hb1);

    if (use_gx) {
        inproj_kernel<<<dim3(64, SEQ), 256, 0, stream>>>(Xe, Wcat, Wb, Gx);
        for (int t = 0; t < SEQ; ++t) {
            const bf16* hin = (t & 1) ? Hb1 : Hb0;
            bf16*      hout = (t & 1) ? Hb0 : Hb1;
            lstm_step_kernel<true><<<128, 128, 0, stream>>>(
                hin, nullptr, Wcat, Wb,
                Gx + (size_t)t * 4 * 256 * 256, C, hout, out);
        }
    } else {
        for (int t = 0; t < SEQ; ++t) {
            const bf16* hin = (t & 1) ? Hb1 : Hb0;
            bf16*      hout = (t & 1) ? Hb0 : Hb1;
            lstm_step_kernel<false><<<128, 128, 0, stream>>>(
                hin, Xe + (size_t)t * NB * EMB, Wcat, Wb,
                nullptr, C, hout, out);
        }
    }
}
